// FlashAttention_60653528154261
// MI455X (gfx1250) — compile-verified, matrix operand order checked by reading, not run on hardware
//
#include <hip/hip_runtime.h>

typedef __bf16 v16bf __attribute__((ext_vector_type(16)));
typedef float  v8f   __attribute__((ext_vector_type(8)));

#define BATCH 4
#define SEQ   4096
#define HDIM  64
#define KPAD  72     // bf16 units: 144B row stride -> 16B aligned, 36-dword bank stagger
#define QPW   32     // queries per wave (two 16-wide N tiles share every A operand)
#define WAVES 4
#define QPB   (QPW * WAVES)   // 128 queries per block
#define NTILE (SEQ / 64)      // 64-key tiles
#define ELEMS ((size_t)BATCH * SEQ * HDIM)

#define ATTN_ASYNC 1          // gfx1250 global_load_async_to_lds_b128 staging

#if __has_builtin(__builtin_amdgcn_exp2f)
#define FAST_EXP2(x) __builtin_amdgcn_exp2f(x)
#else
#define FAST_EXP2(x) exp2f(x)
#endif
#define L2E 1.44269504088896340736f

// fp32 -> bf16 RNE (cold paths: prepass, Q load, fallback staging)
static __device__ __forceinline__ unsigned short f2bf(float f) {
    unsigned u = __builtin_bit_cast(unsigned, f);
    u += 0x7FFFu + ((u >> 16) & 1u);
    return (unsigned short)(u >> 16);
}
static __device__ __forceinline__ unsigned pack2(float a, float b) {
    return (unsigned)f2bf(a) | ((unsigned)f2bf(b) << 16);
}

// hot path: round-half-up to bf16 and pack pair in one v_perm_b32
static __device__ __forceinline__ unsigned pack2_fast(float a, float b) {
    const unsigned ua = __builtin_bit_cast(unsigned, a) + 0x8000u;
    const unsigned ub = __builtin_bit_cast(unsigned, b) + 0x8000u;
#if __has_builtin(__builtin_amdgcn_perm)
    return __builtin_amdgcn_perm(ua, ub, 0x03020706u);  // {ub[31:16], ua[31:16]}
#else
    return (ua >> 16) | (ub & 0xFFFF0000u);
#endif
}

union AB16 { v16bf v; unsigned u[8]; uint4 q[2]; };

// ISA 16-bit A/B layout: VGPR v, lane l -> k_base = (v&4?16:0) + (l&16?8:0) + (v&3)*2
static __device__ __forceinline__ int kbase16(int v, int lane) {
    return ((v & 4) ? 16 : 0) + ((lane & 16) ? 8 : 0) + (v & 3) * 2;
}

// ---------------- prepass: K -> bf16 row-major, V -> bf16 transposed [d][s] -------------
__global__ __launch_bounds__(256) void convert_kv(const float* __restrict__ K,
                                                  const float* __restrict__ V,
                                                  unsigned short* __restrict__ Kbf,
                                                  unsigned short* __restrict__ VT) {
    __shared__ float tile[64][65];
    const int tid = threadIdx.x;
    const int b = blockIdx.y, sblk = blockIdx.x * 64;
    const float* Kb = K + (size_t)b * SEQ * HDIM;
    const float* Vb = V + (size_t)b * SEQ * HDIM;
    unsigned short* Ko = Kbf + (size_t)b * SEQ * HDIM;
    unsigned short* Vo = VT + (size_t)b * HDIM * SEQ;
    for (int i = tid; i < 64 * 64; i += 256) {
        const int s = i >> 6, d = i & 63;
        Ko[(size_t)(sblk + s) * HDIM + d] = f2bf(Kb[(size_t)(sblk + s) * HDIM + d]);
        tile[s][d] = Vb[(size_t)(sblk + s) * HDIM + d];
    }
    __syncthreads();
    for (int i = tid; i < 64 * 64; i += 256) {
        const int d = i >> 6, s = i & 63;
        Vo[(size_t)d * SEQ + sblk + s] = f2bf(tile[s][d]);
    }
}

// ---------------- tile staging: 64 keys of K (row-major) + V^T (row-major) -------------
template <bool USE_WS>
static __device__ __forceinline__ void stage_tile(int tid, int kb,
                                                  const unsigned short* Kbfb,
                                                  const unsigned short* VTb,
                                                  const float* Kfb, const float* Vfb,
                                                  unsigned short* dk, unsigned short* dv) {
    if constexpr (USE_WS) {
#if ATTN_ASYNC
        // 8 async DMA issues per wave per tile; tracked by ASYNCcnt
#pragma unroll
        for (int p = 0; p < 4; ++p) {
            const int c = tid + p * 128;            // 512 x 16B chunks
            const int r = c >> 3, off = (c & 7) * 8;
            const unsigned short* ks = Kbfb + (size_t)(kb + r) * HDIM + off;
            const unsigned short* vs = VTb + (size_t)r * SEQ + kb + off;
            const unsigned dko = (unsigned)(size_t)(dk + r * KPAD + off);
            const unsigned dvo = (unsigned)(size_t)(dv + r * KPAD + off);
            asm volatile("global_load_async_to_lds_b128 %0, %1, off"
                         :: "v"(dko), "v"((unsigned long long)(size_t)ks) : "memory");
            asm volatile("global_load_async_to_lds_b128 %0, %1, off"
                         :: "v"(dvo), "v"((unsigned long long)(size_t)vs) : "memory");
        }
#else
#pragma unroll
        for (int p = 0; p < 4; ++p) {
            const int c = tid + p * 128;
            const int r = c >> 3, off = (c & 7) * 8;
            *(uint4*)(dk + r * KPAD + off) = *(const uint4*)(Kbfb + (size_t)(kb + r) * HDIM + off);
            *(uint4*)(dv + r * KPAD + off) = *(const uint4*)(VTb + (size_t)r * SEQ + kb + off);
        }
#endif
    } else {
        // fallback: convert f32 sources in-kernel (K row-major, V transposed scatter)
#pragma unroll
        for (int p = 0; p < 8; ++p) {
            const int c = tid + p * 128;            // 1024 x float4 chunks
            const int key = c >> 4, off = (c & 15) * 4;
            const float4 kf = *(const float4*)(Kfb + (size_t)(kb + key) * HDIM + off);
            const float4 vf = *(const float4*)(Vfb + (size_t)(kb + key) * HDIM + off);
            uint2 kp; kp.x = pack2(kf.x, kf.y); kp.y = pack2(kf.z, kf.w);
            *(uint2*)(dk + key * KPAD + off) = kp;
            dv[(off + 0) * KPAD + key] = f2bf(vf.x);
            dv[(off + 1) * KPAD + key] = f2bf(vf.y);
            dv[(off + 2) * KPAD + key] = f2bf(vf.z);
            dv[(off + 3) * KPAD + key] = f2bf(vf.w);
        }
    }
}

static __device__ __forceinline__ void wait_async(bool more_inflight) {
#if ATTN_ASYNC
    if (more_inflight) asm volatile("s_wait_asynccnt 0x8" ::: "memory");
    else               asm volatile("s_wait_asynccnt 0x0" ::: "memory");
#endif
}

// ------- main: O^T = V^T * exp2((K)(Q*log2e)^T) over this block's key slice ------------
// blockIdx.z = key-split index; each split accumulates nt tiles and writes its own
// output image at Oout + z*ELEMS (split==1 writes the final output directly).
template <bool USE_WS>
__global__ __launch_bounds__(128) void expattn(const float* __restrict__ Qf,
                                               const float* __restrict__ Kf,
                                               const float* __restrict__ Vf,
                                               const unsigned short* __restrict__ Kbf,
                                               const unsigned short* __restrict__ VT,
                                               float* __restrict__ Oout, int nt) {
    __shared__ __attribute__((aligned(16))) unsigned short lds_k[2][64 * KPAD];
    __shared__ __attribute__((aligned(16))) unsigned short lds_vt[2][64 * KPAD];

    const int tid = threadIdx.x, lane = tid & 31, wave = tid >> 5;
    const int b = blockIdx.y;
    const int qbase = blockIdx.x * QPB + wave * QPW;
    const int sk0 = blockIdx.z * nt;   // first 64-key tile of this split

    const float* Qb = Qf + (size_t)b * SEQ * HDIM;
    const float* Kfb = Kf + (size_t)b * SEQ * HDIM;
    const float* Vfb = Vf + (size_t)b * SEQ * HDIM;
    const unsigned short* Kbfb = USE_WS ? Kbf + (size_t)b * SEQ * HDIM : nullptr;
    const unsigned short* VTb  = USE_WS ? VT  + (size_t)b * HDIM * SEQ : nullptr;
    float* Ob = Oout + (size_t)blockIdx.z * ELEMS + (size_t)b * SEQ * HDIM;

    // Q^T B-operands; log2e folded into fp32 Q before bf16 conversion
    AB16 qa[2][2];
#pragma unroll
    for (int q = 0; q < 2; ++q) {
        const float* qr = Qb + (size_t)(qbase + q * 16 + (lane & 15)) * HDIM;
#pragma unroll
        for (int c = 0; c < 2; ++c)
#pragma unroll
            for (int v = 0; v < 8; ++v) {
                const int d0 = 32 * c + kbase16(v, lane);
                qa[q][c].u[v] = pack2(qr[d0] * L2E, qr[d0 + 1] * L2E);
            }
    }

    v8f acc[2][4];   // O^T tiles: rows = D, lanes = queries
#pragma unroll
    for (int q = 0; q < 2; ++q)
#pragma unroll
        for (int dt = 0; dt < 4; ++dt) acc[q][dt] = (v8f){};

    stage_tile<USE_WS>(tid, sk0 * 64, Kbfb, VTb, Kfb, Vfb, lds_k[0], lds_vt[0]);

    for (int it = 0; it < nt; ++it) {
        const int nb = it + 1;
        if (nb < nt)
            stage_tile<USE_WS>(tid, (sk0 + nb) * 64, Kbfb, VTb, Kfb, Vfb,
                               lds_k[nb & 1], lds_vt[nb & 1]);
        if constexpr (USE_WS) wait_async(nb < nt);
        __syncthreads();

        const unsigned short* lk = lds_k[it & 1];
        const unsigned short* lv = lds_vt[it & 1];

#pragma unroll
        for (int sub = 0; sub < 2; ++sub) {
            AB16 pb[2];   // P^T B-operands (32 keys x 16 queries), built lane-locally
#pragma unroll
            for (int t = 0; t < 2; ++t) {
                const int keyrow = sub * 32 + t * 16 + (lane & 15);
                AB16 ka[2];
#pragma unroll
                for (int c = 0; c < 2; ++c) {
                    const unsigned short* p = &lk[keyrow * KPAD + 32 * c + ((lane & 16) ? 8 : 0)];
                    ka[c].q[0] = *(const uint4*)p;
                    ka[c].q[1] = *(const uint4*)(p + 16);
                }
#pragma unroll
                for (int q = 0; q < 2; ++q) {
                    v8f s = (v8f){};
                    s = __builtin_amdgcn_wmma_f32_16x16x32_bf16(false, ka[0].v, false, qa[q][0].v,
                                                                (short)0, s, false, false);
                    s = __builtin_amdgcn_wmma_f32_16x16x32_bf16(false, ka[1].v, false, qa[q][1].v,
                                                                (short)0, s, false, false);
#pragma unroll
                    for (int j = 0; j < 4; ++j) {
                        const float e0 = FAST_EXP2(s[2 * j]);
                        const float e1 = FAST_EXP2(s[2 * j + 1]);
                        pb[q].u[t * 4 + j] = pack2_fast(e0, e1);
                    }
                }
            }
#pragma unroll
            for (int dt = 0; dt < 4; ++dt) {
                AB16 va;
                const unsigned short* p =
                    &lv[(dt * 16 + (lane & 15)) * KPAD + sub * 32 + ((lane & 16) ? 8 : 0)];
                va.q[0] = *(const uint4*)p;
                va.q[1] = *(const uint4*)(p + 16);
#pragma unroll
                for (int q = 0; q < 2; ++q)
                    acc[q][dt] = __builtin_amdgcn_wmma_f32_16x16x32_bf16(
                        false, va.v, false, pb[q].v, (short)0, acc[q][dt], false, false);
            }
        }
        __syncthreads();
    }

    // O^T C-tile: lane = query, VGPR r = d (contiguous) -> two b128 stores per tile
#pragma unroll
    for (int q = 0; q < 2; ++q) {
        float* orow = Ob + (size_t)(qbase + q * 16 + (lane & 15)) * HDIM + ((lane & 16) ? 8 : 0);
#pragma unroll
        for (int dt = 0; dt < 4; ++dt) {
            float4 o0, o1;
            o0.x = acc[q][dt][0]; o0.y = acc[q][dt][1]; o0.z = acc[q][dt][2]; o0.w = acc[q][dt][3];
            o1.x = acc[q][dt][4]; o1.y = acc[q][dt][5]; o1.z = acc[q][dt][6]; o1.w = acc[q][dt][7];
            *(float4*)(orow + dt * 16)     = o0;
            *(float4*)(orow + dt * 16 + 4) = o1;
        }
    }
}

// ---------------- split-K reduction: out = sum of partial images --------------------
__global__ __launch_bounds__(256) void reduce_split(const float* __restrict__ parts,
                                                    float* __restrict__ out, int nsplit) {
    const size_t i = ((size_t)blockIdx.x * 256 + threadIdx.x) * 4;
    if (i >= ELEMS) return;
    float4 a = *(const float4*)(parts + i);
    for (int sp = 1; sp < nsplit; ++sp) {
        const float4 bq = *(const float4*)(parts + (size_t)sp * ELEMS + i);
        a.x += bq.x; a.y += bq.y; a.z += bq.z; a.w += bq.w;
    }
    *(float4*)(out + i) = a;
}

extern "C" void kernel_launch(void* const* d_in, const int* in_sizes, int n_in,
                              void* d_out, int out_size, void* d_ws, size_t ws_size,
                              hipStream_t stream) {
    const float* q = (const float*)d_in[0];
    const float* k = (const float*)d_in[1];
    const float* v = (const float*)d_in[2];
    float* o = (float*)d_out;

    const size_t bf_bytes = ELEMS * 2ull * sizeof(unsigned short);  // Kbf + VT = 4 MB
    unsigned short* kbf = (unsigned short*)d_ws;
    unsigned short* vt  = kbf + ELEMS;
    float* parts = (float*)((char*)d_ws + bf_bytes);

    // choose largest key-split the workspace allows: 2048/1024/512 waves of work
    int split = 0;
    if (ws_size >= bf_bytes + 4ull * ELEMS * sizeof(float))      split = 4;
    else if (ws_size >= bf_bytes + 2ull * ELEMS * sizeof(float)) split = 2;
    else if (ws_size >= bf_bytes)                                split = 1;

    if (split > 0) {
        convert_kv<<<dim3(SEQ / 64, BATCH), 256, 0, stream>>>(k, v, kbf, vt);
        float* dst = (split == 1) ? o : parts;
        dim3 grid(SEQ / QPB, BATCH, split);
        expattn<true><<<grid, 128, 0, stream>>>(q, k, v, kbf, vt, dst, NTILE / split);
        if (split > 1)
            reduce_split<<<(int)(ELEMS / 4 / 256), 256, 0, stream>>>(parts, o, split);
    } else {
        dim3 grid(SEQ / QPB, BATCH, 1);
        expattn<false><<<grid, 128, 0, stream>>>(q, k, v, nullptr, nullptr, o, NTILE);
    }
}
